// ImprovedCLRN_42726334661046
// MI455X (gfx1250) — matrix operand order repaired and checked by reading, not compiled, not run on hardware
//
#include <hip/hip_runtime.h>
#include <math.h>

typedef __attribute__((ext_vector_type(16))) __bf16 v16bf;
typedef __attribute__((ext_vector_type(8)))  __bf16 v8bf;
typedef __attribute__((ext_vector_type(8)))  float  v8f;
typedef __attribute__((ext_vector_type(4)))  int    v4i;

#ifndef __has_builtin
#define __has_builtin(x) 0
#endif

#if __has_builtin(__builtin_amdgcn_global_load_async_to_lds_b128) && \
    __has_builtin(__builtin_amdgcn_s_wait_asynccnt)
#define USE_ASYNC_LDS 1
#else
#define USE_ASYNC_LDS 0
#endif

// ---------------- workspace layout (bytes) ----------------
#define OFF_W1F   ((size_t)0)                       // bf16 768*256
#define OFF_W2F   (OFF_W1F  + (size_t)768*256*2)    // bf16 256*128
#define OFF_WQF   (OFF_W2F  + (size_t)256*128*2)    // bf16 256*256
#define OFF_WOF   (OFF_WQF  + (size_t)256*256*2)    // bf16 256*256
#define OFF_WVF   (OFF_WOF  + (size_t)256*256*2)    // bf16 8 heads * 256*32
#define OFF_WKTF  (OFF_WVF  + (size_t)256*256*2)    // bf16 8 heads * 32*256
#define OFF_POS   (OFF_WKTF + (size_t)256*256*2)    // f32 32*256 (clipped)
#define OFF_QBK   (OFF_POS  + (size_t)32*256*4)     // f32 4096*8
#define OFF_R     (OFF_QBK  + (size_t)4096*8*4)     // f32 4096*8*256
#define OFF_MIX   (OFF_R    + (size_t)4096*8*256*4) // bf16 4096*8*256

// ---------------- WMMA helpers ----------------
__device__ __forceinline__ v8f wmma_bf16(v16bf a, v16bf b, v8f c) {
  return __builtin_amdgcn_wmma_f32_16x16x32_bf16(false, a, false, b, (short)0, c,
                                                 false, false);
}

// A fragment (16x32 bf16, M=lane&15) from row-major LDS.
// lanes 0-15: k_local = 0..15 ; lanes 16-31: 16..31.
__device__ __forceinline__ v16bf load_a(const __bf16* base, int stride, int m,
                                        int k0, int lane) {
  const __bf16* p = base + m * stride + k0 + ((lane & 16) ? 16 : 0);
  union { v16bf v; v8bf h[2]; } u;
  u.h[0] = *(const v8bf*)(p);
  u.h[1] = *(const v8bf*)(p + 8);
  return u.v;
}

// ---------------- weight repacking into B-fragment layout ----------------
// frag idx = (((s*ntiles + nt)*32 + lane)*16 + e)
// B[k][n] element: k = s*32 + (lane>=16 ? 16:0) + e ; n = nt*16 + (lane&15)
__global__ __launch_bounds__(256) void k_pack(const float* __restrict__ src,
                                              int srcStride, int colOff, int K,
                                              int N, __bf16* __restrict__ dst) {
  int total = K * N;
  int ntiles = N >> 4;
  for (int idx = blockIdx.x * 256 + threadIdx.x; idx < total;
       idx += gridDim.x * 256) {
    int e = idx & 15;
    int lane = (idx >> 4) & 31;
    int rest = idx >> 9;
    int nt = rest % ntiles;
    int s = rest / ntiles;
    int k = s * 32 + ((lane & 16) ? 16 : 0) + e;
    int n = nt * 16 + (lane & 15);
    dst[idx] = (__bf16)src[k * srcStride + colOff + n];
  }
}

// transposed variant: B[k][n] = src[n*srcStride + colOff + k]  (for Wk^T per head)
__global__ __launch_bounds__(256) void k_packT(const float* __restrict__ src,
                                               int srcStride, int colOff, int K,
                                               int N, __bf16* __restrict__ dst) {
  int total = K * N;
  int ntiles = N >> 4;
  for (int idx = blockIdx.x * 256 + threadIdx.x; idx < total;
       idx += gridDim.x * 256) {
    int e = idx & 15;
    int lane = (idx >> 4) & 31;
    int rest = idx >> 9;
    int nt = rest % ntiles;
    int s = rest / ntiles;
    int k = s * 32 + ((lane & 16) ? 16 : 0) + e;
    int n = nt * 16 + (lane & 15);
    dst[idx] = (__bf16)src[n * srcStride + colOff + k];
  }
}

__global__ __launch_bounds__(256) void k_pos(const float* __restrict__ pos,
                                             float* __restrict__ dst) {
  int i = blockIdx.x * 256 + threadIdx.x;
  if (i < 32 * 256) dst[i] = fminf(fmaxf(pos[i], -10.f), 10.f);
}

// ---------------- TGU: gating MLP (per 16-row tile) ----------------
__global__ __launch_bounds__(256) void k_tgu(
    const float* __restrict__ hf, const float* __restrict__ hm,
    const float* __restrict__ hs, const __bf16* __restrict__ W1f,
    const float* __restrict__ b1, const float* __restrict__ lng,
    const float* __restrict__ lnb, const __bf16* __restrict__ W2f,
    const float* __restrict__ b2, const float* __restrict__ W3,
    const float* __restrict__ b3, const float* __restrict__ basew,
    const float* __restrict__ bcoef, const float* __restrict__ acoef,
    const float* __restrict__ Wup, const float* __restrict__ bup,
    float* __restrict__ out) {
  constexpr int XS = 776, HS = 264, H2S = 136;
  __shared__ __bf16 xb[16 * XS];
  __shared__ float hbuf[16 * HS];
  __shared__ __bf16 hb[16 * HS];
  __shared__ float h2[16 * H2S];
  int tid = threadIdx.x, wv = tid >> 5, lane = tid & 31;
  int row0 = blockIdx.x * 16;

  for (int i = tid; i < 16 * 768; i += 256) {
    int r = i / 768, c = i - r * 768;
    float v = (c < 256) ? hf[(row0 + r) * 256 + c]
              : (c < 512) ? hm[(row0 + r) * 256 + c - 256]
                          : hs[(row0 + r) * 256 + c - 512];
    xb[r * XS + c] = (__bf16)v;
  }
  __syncthreads();

  // layer1: (16x768)@(768x256) via WMMA bf16
  for (int nt = wv; nt < 16; nt += 8) {
    v8f acc = {};
    for (int s = 0; s < 24; ++s) {
      v16bf a = load_a(xb, XS, lane & 15, s * 32, lane);
      v16bf bm = *(const v16bf*)(W1f + (((s * 16 + nt) * 32 + lane) << 4));
      acc = wmma_bf16(a, bm, acc);
    }
    int n = nt * 16 + (lane & 15), mb = (lane & 16) ? 8 : 0;
    float bias = b1[n];
#pragma unroll
    for (int r = 0; r < 8; ++r) hbuf[(mb + r) * HS + n] = acc[r] + bias;
  }
  __syncthreads();

  // LayerNorm + ReLU, 2 rows per wave
  for (int rr = 0; rr < 2; ++rr) {
    int r = wv * 2 + rr;
    float s1 = 0.f, s2 = 0.f;
    for (int c = lane; c < 256; c += 32) {
      float v = hbuf[r * HS + c];
      s1 += v;
      s2 += v * v;
    }
    for (int o = 16; o; o >>= 1) {
      s1 += __shfl_xor(s1, o, 32);
      s2 += __shfl_xor(s2, o, 32);
    }
    float mu = s1 * (1.f / 256.f);
    float var = s2 * (1.f / 256.f) - mu * mu;
    float inv = rsqrtf(var + 1e-5f);
    for (int c = lane; c < 256; c += 32) {
      float v = (hbuf[r * HS + c] - mu) * inv * lng[c] + lnb[c];
      hb[r * HS + c] = (__bf16)fmaxf(v, 0.f);
    }
  }
  __syncthreads();

  // layer2: (16x256)@(256x128) + ReLU
  {
    int nt = wv;
    v8f acc = {};
    for (int s = 0; s < 8; ++s) {
      v16bf a = load_a(hb, HS, lane & 15, s * 32, lane);
      v16bf bm = *(const v16bf*)(W2f + (((s * 8 + nt) * 32 + lane) << 4));
      acc = wmma_bf16(a, bm, acc);
    }
    int n = nt * 16 + (lane & 15), mb = (lane & 16) ? 8 : 0;
    float bias = b2[n];
#pragma unroll
    for (int r = 0; r < 8; ++r) h2[(mb + r) * H2S + n] = fmaxf(acc[r] + bias, 0.f);
  }
  __syncthreads();

  if (tid < 16) {  // layer3 + gating combine per row
    int r = tid;
    float lg0 = b3[0], lg1 = b3[1], lg2 = b3[2];
    for (int c = 0; c < 128; ++c) {
      float v = h2[r * H2S + c];
      lg0 += v * W3[c * 3 + 0];
      lg1 += v * W3[c * 3 + 1];
      lg2 += v * W3[c * 3 + 2];
    }
    float mx = fmaxf(lg0, fmaxf(lg1, lg2));
    float e0 = expf(lg0 - mx), e1 = expf(lg1 - mx), e2 = expf(lg2 - mx);
    float inv = 1.f / (e0 + e1 + e2);
    float a0 = e0 * inv, a1 = e1 * inv, a2 = e2 * inv;
    float bw0 = basew[0], bw1 = basew[1], bw2 = basew[2];
    float bmx = fmaxf(bw0, fmaxf(bw1, bw2));
    float f0 = expf(bw0 - bmx), f1 = expf(bw1 - bmx), f2 = expf(bw2 - bmx);
    float binv = 1.f / (f0 + f1 + f2);
    float cb = 1.f / (1.f + expf(-bcoef[0]));
    float ca = 1.f / (1.f + expf(-acoef[0]));
    float tot = cb + ca, wb = cb / tot, wa = ca / tot;
    float c0 = wb * f0 * binv + wa * a0;
    float c1 = wb * f1 * binv + wa * a1;
    float c2 = wb * f2 * binv + wa * a2;
    float cmx = fmaxf(c0, fmaxf(c1, c2));
    float g0 = expf(c0 - cmx), g1 = expf(c1 - cmx), g2 = expf(c2 - cmx);
    float ginv = 1.f / (g0 + g1 + g2);
    float* po = out + (size_t)(row0 + r) * 261 + 256;
    po[0] = g0 * ginv;
    po[1] = g1 * ginv;
    po[2] = g2 * ginv;
  } else if (tid < 32) {  // update_probs = sigmoid(x@Wup + bup)
    int r = tid - 16;
    float u0 = bup[0], u1 = bup[1];
    for (int c = 0; c < 768; ++c) {
      float x = (float)xb[r * XS + c];
      u0 += x * Wup[c * 2 + 0];
      u1 += x * Wup[c * 2 + 1];
    }
    float* po = out + (size_t)(row0 + r) * 261 + 259;
    po[0] = 1.f / (1.f + expf(-u0));
    po[1] = 1.f / (1.f + expf(-u1));
  }
}

// ---------------- q projection + R = Wk_h @ q_h + q.bk ----------------
__global__ __launch_bounds__(256) void k_qr(
    const float* __restrict__ query, const __bf16* __restrict__ Wqf,
    const float* __restrict__ bq, const float* __restrict__ bk,
    const __bf16* __restrict__ Wktf, float* __restrict__ Rws,
    float* __restrict__ qbkws) {
  constexpr int QS = 264;
  __shared__ __bf16 s_qa[16 * QS];
  __shared__ float s_q[16 * QS];
  int tid = threadIdx.x, wv = tid >> 5, lane = tid & 31;
  int row0 = blockIdx.x * 16;

  for (int i = tid; i < 16 * 256; i += 256) {
    int r = i >> 8, c = i & 255;
    s_qa[r * QS + c] = (__bf16)query[(row0 + r) * 256 + c];
  }
  __syncthreads();

  for (int nt = wv; nt < 16; nt += 8) {  // q = query@Wq + bq
    v8f acc = {};
    for (int s = 0; s < 8; ++s) {
      v16bf a = load_a(s_qa, QS, lane & 15, s * 32, lane);
      v16bf bm = *(const v16bf*)(Wqf + (((s * 16 + nt) * 32 + lane) << 4));
      acc = wmma_bf16(a, bm, acc);
    }
    int n = nt * 16 + (lane & 15), mb = (lane & 16) ? 8 : 0;
    float bias = bq[n];
#pragma unroll
    for (int r = 0; r < 8; ++r) s_q[(mb + r) * QS + n] = acc[r] + bias;
  }
  __syncthreads();

  if (tid < 128) {  // qbk[b][h] = q_head . bk_head
    int r = tid >> 3, h = tid & 7;
    float s = 0.f;
    for (int j = 0; j < 32; ++j) s += s_q[r * QS + h * 32 + j] * bk[h * 32 + j];
    qbkws[(size_t)(row0 + r) * 8 + h] = s;
  }
  for (int i = tid; i < 16 * 256; i += 256) {
    int r = i >> 8, c = i & 255;
    s_qa[r * QS + c] = (__bf16)s_q[r * QS + c];
  }
  __syncthreads();

  // R[b,h,:] : per (h,nt) a single K=32 WMMA with A = q head slice
  for (int p = wv; p < 128; p += 8) {
    int h = p >> 4, nt = p & 15;
    v16bf a = load_a(s_qa, QS, lane & 15, h * 32, lane);
    v16bf bm = *(const v16bf*)(Wktf + h * 8192 + (((nt) * 32 + lane) << 4));
    v8f acc = {};
    acc = wmma_bf16(a, bm, acc);
    int n = nt * 16 + (lane & 15), mb = (lane & 16) ? 8 : 0;
#pragma unroll
    for (int r = 0; r < 8; ++r)
      Rws[((size_t)(row0 + mb + r) * 8 + h) * 256 + n] = acc[r];
  }
}

// ---------------- fused attention core: one block per batch row ----------------
// keys tile goes memory -> LDS via GLOBAL_LOAD_ASYNC_TO_LDS_B128 (ASYNCcnt path)
// when the toolchain exposes it; values tile uses the bf16 register path.
__global__ __launch_bounds__(256) void k_attn(
    const float* __restrict__ keys, const float* __restrict__ vals,
    const unsigned char* __restrict__ mask, const float* __restrict__ posf,
    const float* __restrict__ Rws, const float* __restrict__ qbkws,
    __bf16* __restrict__ mixws) {
  constexpr int KS = 260;  // f32 keys stride (1040 B rows: 16B-aligned, 2-way bank)
  constexpr int VS = 268;  // bf16 values stride (conflict-free column reads)
  constexpr int RS = 264;
  __shared__ float kf[32 * KS];    // 33280 B, raw f32 keys
  __shared__ __bf16 vls[32 * VS];  // 17152 B
  __shared__ float Rls[8 * RS];    // 8448 B
  __shared__ float attn[8 * 32];   // 1024 B
  int b = blockIdx.x, tid = threadIdx.x, wv = tid >> 5, lane = tid & 31;

#if USE_ASYNC_LDS
  // 32 rows * 1024 B; one async B128 moves 512 B per wave-issue (16 B/lane).
  for (int c = wv; c < 64; c += 8) {
    int t = c >> 1, half = c & 1;
    const float* gp = keys + ((size_t)t * 4096 + b) * 256 + half * 128 + lane * 4;
    float* lp = kf + t * KS + half * 128 + lane * 4;
    __builtin_amdgcn_global_load_async_to_lds_b128(
        (__attribute__((address_space(1))) v4i*)gp,
        (__attribute__((address_space(3))) v4i*)lp, 0, 0);
  }
#else
  for (int i = tid; i < 32 * 256; i += 256) {
    int t = i >> 8, d = i & 255;
    kf[t * KS + d] = keys[((size_t)t * 4096 + b) * 256 + d];
  }
#endif
  for (int i = tid; i < 32 * 256; i += 256) {
    int t = i >> 8, d = i & 255;
    vls[t * VS + d] = (__bf16)vals[((size_t)t * 4096 + b) * 256 + d];
  }
  for (int i = tid; i < 8 * 256; i += 256) {
    int h = i >> 8, d = i & 255;
    Rls[h * RS + d] = Rws[((size_t)b * 8 + h) * 256 + d];
  }
#if USE_ASYNC_LDS
  __builtin_amdgcn_s_wait_asynccnt(0);
#endif
  __syncthreads();

  // keys += clipped positional encoding (posf is 32 KB, L2-resident)
  for (int i = tid; i < 32 * 256; i += 256) {
    int t = i >> 8, d = i & 255;
    kf[t * KS + d] += posf[i];
  }
  __syncthreads();

  {  // scores + wave softmax: wave = head, lane = t
    int h = wv, t = lane;
    float s = 0.f;
    for (int d = 0; d < 256; ++d) s += kf[t * KS + d] * Rls[h * RS + d];
    s = (s + qbkws[(size_t)b * 8 + h]) * 0.1767766952966369f;  // 1/sqrt(32)
    s = fminf(fmaxf(s, -50.f), 50.f);
    if (!mask[b * 32 + t]) s = -INFINITY;
    float mx = s;
    for (int o = 16; o; o >>= 1) mx = fmaxf(mx, __shfl_xor(mx, o, 32));
    float e = expf(s - mx), sum = e;
    for (int o = 16; o; o >>= 1) sum += __shfl_xor(sum, o, 32);
    attn[h * 32 + t] = e / sum;
  }
  __syncthreads();

  {  // mix[b,h,:] = attn @ values_tile
    int h = wv;
#pragma unroll
    for (int i = 0; i < 8; ++i) {
      int d = i * 32 + lane;
      float acc = 0.f;
      for (int t = 0; t < 32; ++t) acc += attn[h * 32 + t] * (float)vls[t * VS + d];
      mixws[((size_t)b * 8 + h) * 256 + d] = (__bf16)acc;
    }
  }
}

// ---------------- output: per-head mix@Wv, then @Wo + bo ----------------
__global__ __launch_bounds__(256) void k_out(const __bf16* __restrict__ mixws,
                                             const __bf16* __restrict__ Wvf,
                                             const float* __restrict__ bv,
                                             const __bf16* __restrict__ Wof,
                                             const float* __restrict__ bo,
                                             float* __restrict__ out) {
  constexpr int MS = 1032, OS = 264;
  __shared__ __bf16 mixb[16 * MS];  // 4 heads per pass
  __shared__ __bf16 ob[16 * OS];
  int tid = threadIdx.x, wv = tid >> 5, lane = tid & 31;
  int row0 = blockIdx.x * 16;

  for (int pass = 0; pass < 2; ++pass) {
    for (int i = tid; i < 16 * 1024; i += 256) {
      int r = i >> 10, c = i & 1023;
      mixb[r * MS + c] = mixws[(size_t)(row0 + r) * 2048 + pass * 1024 + c];
    }
    __syncthreads();
    {  // 8 (head, ntile) pairs this pass, one per wave
      int p = wv;
      int hl = p >> 1;        // local head 0..3
      int h = pass * 4 + hl;  // global head
      int nt = p & 1;
      v8f acc = {};
      for (int s = 0; s < 8; ++s) {
        v16bf a = load_a(mixb, MS, lane & 15, hl * 256 + s * 32, lane);
        v16bf bm =
            *(const v16bf*)(Wvf + h * 8192 + (((s * 2 + nt) * 32 + lane) << 4));
        acc = wmma_bf16(a, bm, acc);
      }
      int n = h * 32 + nt * 16 + (lane & 15), mb = (lane & 16) ? 8 : 0;
      float bias = bv[n];
#pragma unroll
      for (int r = 0; r < 8; ++r) ob[(mb + r) * OS + n] = (__bf16)(acc[r] + bias);
    }
    __syncthreads();
  }

  for (int nt = wv; nt < 16; nt += 8) {  // attended = out16 @ Wo + bo
    v8f acc = {};
    for (int s = 0; s < 8; ++s) {
      v16bf a = load_a(ob, OS, lane & 15, s * 32, lane);
      v16bf bm = *(const v16bf*)(Wof + (((s * 16 + nt) * 32 + lane) << 4));
      acc = wmma_bf16(a, bm, acc);
    }
    int n = nt * 16 + (lane & 15), mb = (lane & 16) ? 8 : 0;
    float bias = bo[n];
#pragma unroll
    for (int r = 0; r < 8; ++r)
      out[(size_t)(row0 + mb + r) * 261 + n] = acc[r] + bias;
  }
}

// ---------------- host launch ----------------
extern "C" void kernel_launch(void* const* d_in, const int* in_sizes, int n_in,
                              void* d_out, int out_size, void* d_ws,
                              size_t ws_size, hipStream_t stream) {
  (void)in_sizes; (void)n_in; (void)out_size; (void)ws_size;
  const float* h_fast = (const float*)d_in[0];
  const float* h_med  = (const float*)d_in[1];
  const float* h_slow = (const float*)d_in[2];
  const float* query  = (const float*)d_in[3];
  const float* keys   = (const float*)d_in[4];
  const float* values = (const float*)d_in[5];
  const unsigned char* mask = (const unsigned char*)d_in[6];
  const float* W1 = (const float*)d_in[7];  const float* b1 = (const float*)d_in[8];
  const float* ln_g = (const float*)d_in[9]; const float* ln_b = (const float*)d_in[10];
  const float* W2 = (const float*)d_in[11]; const float* b2 = (const float*)d_in[12];
  const float* W3 = (const float*)d_in[13]; const float* b3 = (const float*)d_in[14];
  const float* basew = (const float*)d_in[15];
  const float* bcoef = (const float*)d_in[16];
  const float* acoef = (const float*)d_in[17];
  const float* Wup = (const float*)d_in[18]; const float* bup = (const float*)d_in[19];
  const float* Wq = (const float*)d_in[20];  const float* bq = (const float*)d_in[21];
  const float* Wk = (const float*)d_in[22];  const float* bk = (const float*)d_in[23];
  const float* Wv = (const float*)d_in[24];  const float* bv = (const float*)d_in[25];
  const float* Wo = (const float*)d_in[26];  const float* bo = (const float*)d_in[27];
  const float* pos = (const float*)d_in[28];

  char* ws = (char*)d_ws;
  __bf16* W1f  = (__bf16*)(ws + OFF_W1F);
  __bf16* W2f  = (__bf16*)(ws + OFF_W2F);
  __bf16* Wqf  = (__bf16*)(ws + OFF_WQF);
  __bf16* Wof  = (__bf16*)(ws + OFF_WOF);
  __bf16* Wvf  = (__bf16*)(ws + OFF_WVF);
  __bf16* Wktf = (__bf16*)(ws + OFF_WKTF);
  float*  posf = (float*)(ws + OFF_POS);
  float*  qbkf = (float*)(ws + OFF_QBK);
  float*  Rf   = (float*)(ws + OFF_R);
  __bf16* mixf = (__bf16*)(ws + OFF_MIX);
  float* out = (float*)d_out;

  k_pack<<<768, 256, 0, stream>>>(W1, 256, 0, 768, 256, W1f);
  k_pack<<<128, 256, 0, stream>>>(W2, 128, 0, 256, 128, W2f);
  k_pack<<<256, 256, 0, stream>>>(Wq, 256, 0, 256, 256, Wqf);
  k_pack<<<256, 256, 0, stream>>>(Wo, 256, 0, 256, 256, Wof);
  for (int h = 0; h < 8; ++h) {
    k_pack<<<32, 256, 0, stream>>>(Wv, 256, h * 32, 256, 32, Wvf + h * 8192);
    k_packT<<<32, 256, 0, stream>>>(Wk, 256, h * 32, 32, 256, Wktf + h * 8192);
  }
  k_pos<<<32, 256, 0, stream>>>(pos, posf);

  k_tgu<<<256, 256, 0, stream>>>(h_fast, h_med, h_slow, W1f, b1, ln_g, ln_b,
                                 W2f, b2, W3, b3, basew, bcoef, acoef, Wup, bup,
                                 out);
  k_qr<<<256, 256, 0, stream>>>(query, Wqf, bq, bk, Wktf, Rf, qbkf);
  k_attn<<<4096, 256, 0, stream>>>(keys, values, mask, posf, Rf, qbkf, mixf);
  k_out<<<256, 256, 0, stream>>>(mixf, Wvf, bv, Wof, bo, out);
}
